// MultiHeadAttention_7387343749848
// MI455X (gfx1250) — matrix operand order repaired and checked by reading, not compiled, not run on hardware
//
#include <hip/hip_runtime.h>
#include <hip/hip_bf16.h>
#include <math.h>

// ---- problem constants (match reference) ----
#define BATCH   2
#define NHEAD   16
#define LQ      2048
#define LK      2048
#define DMODEL  1024
#define DHEAD   64
#define NTOK    (BATCH * LQ)          // 4096 rows
#define LN_EPS  1e-6f

typedef __bf16 bf16;
typedef __attribute__((ext_vector_type(16))) __bf16 v16bf;
typedef __attribute__((ext_vector_type(8)))  __bf16 v8bf;
typedef __attribute__((ext_vector_type(8)))  float  v8f;

// ---------------------------------------------------------------------------
// Fragment loaders for v_wmma_f32_16x16x32_bf16 (wave32).
// A (16x32, MxK): lane L<16 -> M=L, f[0..7]=K0..7, f[8..15]=K16..23
//                 lane L>=16 -> M=L-16, f[0..7]=K8..15, f[8..15]=K24..31
// B (32x16, KxN): lane L<16 -> N=L, f[0..7]=K0..7, f[8..15]=K16..23
//                 lane L>=16 -> N=L-16, f[0..7]=K8..15, f[8..15]=K24..31
// Sources are row-major [own-dim][K], K contiguous -> pure b128 loads.
// ---------------------------------------------------------------------------
__device__ __forceinline__ v16bf load_frag_a(const bf16* base, int row0, int ld, int k0) {
    const int lane = threadIdx.x & 31;
    const int r  = lane & 15;
    const int hi = lane >> 4;
    const bf16* p = base + (size_t)(row0 + r) * ld + k0 + hi * 8;
    v8bf lo = *(const v8bf*)(p);
    v8bf hh = *(const v8bf*)(p + 16);
    v16bf f;
#pragma unroll
    for (int i = 0; i < 8; ++i) { f[i] = lo[i]; f[8 + i] = hh[i]; }
    return f;
}

__device__ __forceinline__ v16bf load_frag_b(const bf16* base, int row0, int ld, int k0) {
    const int lane = threadIdx.x & 31;
    const int n  = lane & 15;
    const int hi = lane >> 4;
    const bf16* p = base + (size_t)(row0 + n) * ld + k0 + hi * 8;
    v8bf lo = *(const v8bf*)(p);
    v8bf hh = *(const v8bf*)(p + 16);
    v16bf f;
#pragma unroll
    for (int i = 0; i < 8; ++i) { f[i] = lo[i]; f[8 + i] = hh[i]; }
    return f;
}

#define WMMA_BF16(a, b, c) \
    __builtin_amdgcn_wmma_f32_16x16x32_bf16(false, (a), false, (b), (short)0, (c), false, false)

// ---------------------------------------------------------------------------
// Elementwise f32 -> bf16 conversion
// ---------------------------------------------------------------------------
__global__ void k_f32_to_bf16(const float* __restrict__ in, bf16* __restrict__ out, int n) {
    int i = blockIdx.x * 256 + threadIdx.x;
    if (i < n) out[i] = (bf16)in[i];
}

// ---------------------------------------------------------------------------
// Projection GEMM: Y = X @ W^T  (X:[NTOK,DMODEL] bf16, W:[DMODEL,DMODEL] bf16)
// One wave computes a 16x64 output strip, register double-buffered over K.
// mode 0: Q (scaled 1/8, [B,H,L,DH])  mode 1: K ([B,H,L,DH])
// mode 2: V transposed ([B,H,DH,L])
// ---------------------------------------------------------------------------
__global__ void k_proj_gemm(const bf16* __restrict__ X, const bf16* __restrict__ W,
                            bf16* __restrict__ dst, int mode) {
    const int nstrips = DMODEL / 64;                 // 16
    const int m0 = (blockIdx.x / nstrips) * 16;
    const int n0 = (blockIdx.x % nstrips) * 64;
    const int lane = threadIdx.x & 31;
    const int n_ = lane & 15, hi = lane >> 4;

    v8f acc[4];
#pragma unroll
    for (int t = 0; t < 4; ++t)
#pragma unroll
        for (int i = 0; i < 8; ++i) acc[t][i] = 0.0f;

    // software pipeline: fragments for k0 in flight while k0-32 computes
    v16bf a = load_frag_a(X, m0, DMODEL, 0);
    v16bf b[4];
#pragma unroll
    for (int t = 0; t < 4; ++t) b[t] = load_frag_b(W, n0 + t * 16, DMODEL, 0);

    for (int k0 = 0; k0 < DMODEL; k0 += 32) {
        const int kn = (k0 + 32) & (DMODEL - 1);     // wraps to 0 on last iter
        v16bf an = load_frag_a(X, m0, DMODEL, kn);
        v16bf bn[4];
#pragma unroll
        for (int t = 0; t < 4; ++t) bn[t] = load_frag_b(W, n0 + t * 16, DMODEL, kn);
#pragma unroll
        for (int t = 0; t < 4; ++t) acc[t] = WMMA_BF16(a, b[t], acc[t]);
        a = an;
#pragma unroll
        for (int t = 0; t < 4; ++t) b[t] = bn[t];
    }

#pragma unroll
    for (int t = 0; t < 4; ++t) {
#pragma unroll
        for (int r = 0; r < 8; ++r) {
            float v = acc[t][r];
            int mg = m0 + r + hi * 8;                // global row (token)
            int ng = n0 + t * 16 + n_;               // global col (feature)
            int bb = mg >> 11;                       // / LQ
            int l  = mg & (LQ - 1);
            int h  = ng >> 6;                        // / DHEAD
            int d  = ng & (DHEAD - 1);
            if (mode == 0) {
                v *= 0.125f;                         // 1/sqrt(DHEAD)
                dst[(((size_t)(bb * NHEAD + h) * LQ) + l) * DHEAD + d] = (bf16)v;
            } else if (mode == 1) {
                dst[(((size_t)(bb * NHEAD + h) * LK) + l) * DHEAD + d] = (bf16)v;
            } else {
                dst[(((size_t)(bb * NHEAD + h) * DHEAD) + d) * LK + l] = (bf16)v;
            }
        }
    }
}

// ---------------------------------------------------------------------------
// Attention pass 1: raw scores S = (Q/temp)·K^T + bias (masked) into attn
// region (scratch, non-temporal), with online per-row max/sum stats.
// Grid: B*H*(LQ/16) waves; each wave: 16 q-rows x full LK.
// ---------------------------------------------------------------------------
__global__ void k_att_scores(const bf16* __restrict__ qh, const bf16* __restrict__ kh,
                             const float* __restrict__ bias, const int* __restrict__ mask,
                             float* __restrict__ rawS, float* __restrict__ stats) {
    const int qtiles = LQ / 16;                      // 128
    const int q0 = (blockIdx.x % qtiles) * 16;
    const int bh = blockIdx.x / qtiles;              // b*NHEAD + h
    const int b  = bh >> 4;
    const int lane = threadIdx.x & 31;
    const int n_ = lane & 15, hi = lane >> 4;

    const bf16* qb = qh + (size_t)bh * LQ * DHEAD;
    const bf16* kb = kh + (size_t)bh * LK * DHEAD;

    v16bf a0 = load_frag_a(qb, q0, DHEAD, 0);
    v16bf a1 = load_frag_a(qb, q0, DHEAD, 32);

    float mrun[8], lrun[8];
#pragma unroll
    for (int r = 0; r < 8; ++r) { mrun[r] = -1e30f; lrun[r] = 0.0f; }

    v16bf b0 = load_frag_b(kb, 0, DHEAD, 0);
    v16bf b1 = load_frag_b(kb, 0, DHEAD, 32);

    for (int k0 = 0; k0 < LK; k0 += 16) {
        const int kn = (k0 + 16) & (LK - 1);
        // prefetch next bias/mask tile (16 rows x 64B): each half-wave covers
        // all 16 rows with one global_prefetch per lane
        {
            size_t pidx = ((size_t)b * LQ + q0 + n_) * LK + kn;
            if (hi == 0) __builtin_prefetch(&bias[pidx], 0, 1);
            else         __builtin_prefetch(&mask[pidx], 0, 1);
        }
        v16bf b0n = load_frag_b(kb, kn, DHEAD, 0);
        v16bf b1n = load_frag_b(kb, kn, DHEAD, 32);
        v8f c = {};
        c = WMMA_BF16(a0, b0, c);
        c = WMMA_BF16(a1, b1, c);
#pragma unroll
        for (int r = 0; r < 8; ++r) {
            int mg = q0 + r + hi * 8;                // q row within LQ
            int ng = k0 + n_;                        // key within LK
            size_t bidx = ((size_t)b * LQ + mg) * LK + ng;
            int   mk = __builtin_nontemporal_load(&mask[bidx]);
            float pb = __builtin_nontemporal_load(&bias[bidx]);
            float s  = (mk == 0) ? -1e9f : (c[r] + pb);
            __builtin_nontemporal_store(s, &rawS[((size_t)bh * LQ + mg) * LK + ng]);
            // row max over the 16 lanes of this half
            float t = s;
            t = fmaxf(t, __shfl_xor(t, 1, 32));
            t = fmaxf(t, __shfl_xor(t, 2, 32));
            t = fmaxf(t, __shfl_xor(t, 4, 32));
            t = fmaxf(t, __shfl_xor(t, 8, 32));
            float nm = fmaxf(mrun[r], t);
            float e = __expf(s - nm);
            e += __shfl_xor(e, 1, 32);
            e += __shfl_xor(e, 2, 32);
            e += __shfl_xor(e, 4, 32);
            e += __shfl_xor(e, 8, 32);
            lrun[r] = lrun[r] * __expf(mrun[r] - nm) + e;
            mrun[r] = nm;
        }
        b0 = b0n; b1 = b1n;
    }
    if (n_ == 0) {
#pragma unroll
        for (int r = 0; r < 8; ++r) {
            size_t row = (size_t)bh * LQ + q0 + r + hi * 8;
            stats[row * 2 + 0] = mrun[r];
            stats[row * 2 + 1] = lrun[r];
        }
    }
}

// ---------------------------------------------------------------------------
// Attention pass 2: attn = exp(S-m)/l (written back over rawS, non-temporal),
// then O = attn @ V via WMMA (P transposed C-layout -> A-layout through LDS).
// ---------------------------------------------------------------------------
__global__ void k_att_pv(const bf16* __restrict__ vT, float* __restrict__ attn,
                         const float* __restrict__ stats, bf16* __restrict__ head_out) {
    const int qtiles = LQ / 16;
    const int q0 = (blockIdx.x % qtiles) * 16;
    const int bh = blockIdx.x / qtiles;
    const int b  = bh >> 4;
    const int h  = bh & 15;
    const int lane = threadIdx.x & 31;
    const int n_ = lane & 15, hi = lane >> 4;

    __shared__ bf16 ldsP[16 * 32];

    float m_i[8], inv_l[8];
#pragma unroll
    for (int r = 0; r < 8; ++r) {
        size_t row = (size_t)bh * LQ + q0 + r + hi * 8;
        m_i[r]  = stats[row * 2 + 0];
        float l = stats[row * 2 + 1];
        inv_l[r] = 1.0f / l;
    }

    const bf16* vb = vT + (size_t)bh * DHEAD * LK;   // [DHEAD][LK]

    v8f o[4];
#pragma unroll
    for (int t = 0; t < 4; ++t)
#pragma unroll
        for (int i = 0; i < 8; ++i) o[t][i] = 0.0f;

    // V fragments double-buffered over the k loop
    v16bf bv[4];
#pragma unroll
    for (int t = 0; t < 4; ++t) bv[t] = load_frag_b(vb, t * 16, LK, 0);

    for (int k0 = 0; k0 < LK; k0 += 32) {
        const int kn = (k0 + 32) & (LK - 1);
        v16bf bvn[4];
#pragma unroll
        for (int t = 0; t < 4; ++t) bvn[t] = load_frag_b(vb, t * 16, LK, kn);
        // prefetch next rawS tile (16 rows x 128B -> two prefetches per row)
        {
            size_t pidx = ((size_t)bh * LQ + q0 + n_) * LK + kn + hi * 16;
            __builtin_prefetch(&attn[pidx], 0, 1);
        }
#pragma unroll
        for (int ks = 0; ks < 32; ks += 16) {
#pragma unroll
            for (int r = 0; r < 8; ++r) {
                int mg = q0 + r + hi * 8;
                int ng = k0 + ks + n_;
                size_t idx = ((size_t)bh * LQ + mg) * LK + ng;
                float s = __builtin_nontemporal_load(&attn[idx]);
                float p = __expf(s - m_i[r]) * inv_l[r];
                __builtin_nontemporal_store(p, &attn[idx]);   // final attn
                ldsP[(r + hi * 8) * 32 + ks + n_] = (bf16)p;
            }
        }
        __syncthreads();
        v16bf a = load_frag_a(ldsP, 0, 32, 0);       // P tile 16(q) x 32(k)
#pragma unroll
        for (int t = 0; t < 4; ++t) o[t] = WMMA_BF16(a, bv[t], o[t]);
        __syncthreads();
#pragma unroll
        for (int t = 0; t < 4; ++t) bv[t] = bvn[t];
    }

#pragma unroll
    for (int t = 0; t < 4; ++t) {
#pragma unroll
        for (int r = 0; r < 8; ++r) {
            int mg  = q0 + r + hi * 8;
            int col = h * DHEAD + t * 16 + n_;
            head_out[((size_t)b * LQ + mg) * DMODEL + col] = (bf16)o[t][r];
        }
    }
}

// ---------------------------------------------------------------------------
// Output projection: pre = head_out @ Wo^T + residual(q), f32 out
// ---------------------------------------------------------------------------
__global__ void k_out_gemm(const bf16* __restrict__ Hh, const bf16* __restrict__ Wo,
                           const float* __restrict__ resid, float* __restrict__ out_pre) {
    const int nstrips = DMODEL / 64;
    const int m0 = (blockIdx.x / nstrips) * 16;
    const int n0 = (blockIdx.x % nstrips) * 64;
    const int lane = threadIdx.x & 31;
    const int n_ = lane & 15, hi = lane >> 4;

    v8f acc[4];
#pragma unroll
    for (int t = 0; t < 4; ++t)
#pragma unroll
        for (int i = 0; i < 8; ++i) acc[t][i] = 0.0f;

    v16bf a = load_frag_a(Hh, m0, DMODEL, 0);
    v16bf b[4];
#pragma unroll
    for (int t = 0; t < 4; ++t) b[t] = load_frag_b(Wo, n0 + t * 16, DMODEL, 0);

    for (int k0 = 0; k0 < DMODEL; k0 += 32) {
        const int kn = (k0 + 32) & (DMODEL - 1);
        v16bf an = load_frag_a(Hh, m0, DMODEL, kn);
        v16bf bn[4];
#pragma unroll
        for (int t = 0; t < 4; ++t) bn[t] = load_frag_b(Wo, n0 + t * 16, DMODEL, kn);
#pragma unroll
        for (int t = 0; t < 4; ++t) acc[t] = WMMA_BF16(a, b[t], acc[t]);
        a = an;
#pragma unroll
        for (int t = 0; t < 4; ++t) b[t] = bn[t];
    }

#pragma unroll
    for (int t = 0; t < 4; ++t) {
#pragma unroll
        for (int r = 0; r < 8; ++r) {
            int mg = m0 + r + hi * 8;
            int ng = n0 + t * 16 + n_;
            size_t idx = (size_t)mg * DMODEL + ng;
            out_pre[idx] = acc[t][r] + resid[idx];
        }
    }
}

// ---------------------------------------------------------------------------
// LayerNorm over last dim (1024), one 256-thread block per row
// ---------------------------------------------------------------------------
__global__ void k_layernorm(const float* __restrict__ x, const float* __restrict__ gamma,
                            const float* __restrict__ beta, float* __restrict__ out) {
    const int row = blockIdx.x;
    const int tid = threadIdx.x;
    __shared__ float s1[256], s2[256];
    float a = 0.0f, sq = 0.0f;
    for (int c = tid; c < DMODEL; c += 256) {
        float v = x[(size_t)row * DMODEL + c];
        a += v; sq += v * v;
    }
    s1[tid] = a; s2[tid] = sq;
    __syncthreads();
    for (int off = 128; off > 0; off >>= 1) {
        if (tid < off) { s1[tid] += s1[tid + off]; s2[tid] += s2[tid + off]; }
        __syncthreads();
    }
    float mu  = s1[0] * (1.0f / DMODEL);
    float var = s2[0] * (1.0f / DMODEL) - mu * mu;
    float rs  = rsqrtf(var + LN_EPS);
    for (int c = tid; c < DMODEL; c += 256) {
        float v = x[(size_t)row * DMODEL + c];
        out[(size_t)row * DMODEL + c] = (v - mu) * rs * gamma[c] + beta[c];
    }
}

// ---------------------------------------------------------------------------
// Host launcher
// ---------------------------------------------------------------------------
extern "C" void kernel_launch(void* const* d_in, const int* in_sizes, int n_in,
                              void* d_out, int out_size, void* d_ws, size_t ws_size,
                              hipStream_t stream) {
    const float* q_in   = (const float*)d_in[0];
    const float* k_in   = (const float*)d_in[1];
    const float* v_in   = (const float*)d_in[2];
    const int*   mask   = (const int*)  d_in[3];
    const float* pbias  = (const float*)d_in[4];
    const float* Wq     = (const float*)d_in[5];
    const float* Wk     = (const float*)d_in[6];
    const float* Wv     = (const float*)d_in[7];
    const float* Wo     = (const float*)d_in[8];
    const float* gamma  = (const float*)d_in[9];
    const float* beta   = (const float*)d_in[10];

    float* out  = (float*)d_out;                          // [B, LQ, DMODEL]
    float* attn = out + (size_t)NTOK * DMODEL;            // [B, H, LQ, LK]

    // ---- workspace carve-up ----
    char* w = (char*)d_ws;
    const size_t ACT = (size_t)NTOK * DMODEL;             // 4,194,304 elems
    const size_t WEL = (size_t)DMODEL * DMODEL;           // 1,048,576 elems
    bf16* qbf = (bf16*)w;                 w += ACT * sizeof(bf16);
    bf16* kbf = (bf16*)w;                 w += ACT * sizeof(bf16);
    bf16* vbf = (bf16*)w;                 w += ACT * sizeof(bf16);
    bf16* wqb = (bf16*)w;                 w += WEL * sizeof(bf16);
    bf16* wkb = (bf16*)w;                 w += WEL * sizeof(bf16);
    bf16* wvb = (bf16*)w;                 w += WEL * sizeof(bf16);
    bf16* wob = (bf16*)w;                 w += WEL * sizeof(bf16);
    bf16* qh  = (bf16*)w;                 w += ACT * sizeof(bf16);   // [B,H,LQ,DH]
    bf16* kh  = (bf16*)w;                 w += ACT * sizeof(bf16);   // [B,H,LK,DH]
    bf16* vT  = (bf16*)w;                 w += ACT * sizeof(bf16);   // [B,H,DH,LK]
    bf16* ho  = (bf16*)w;                 w += ACT * sizeof(bf16);   // [B,LQ,DMODEL]
    float* stats = (float*)w;             w += (size_t)BATCH * NHEAD * LQ * 2 * sizeof(float);
    float* pre   = (float*)w;             w += ACT * sizeof(float);
    (void)ws_size; (void)in_sizes; (void)n_in; (void)out_size;

    // 1) f32 -> bf16 conversions
    k_f32_to_bf16<<<(ACT + 255) / 256, 256, 0, stream>>>(q_in, qbf, (int)ACT);
    k_f32_to_bf16<<<(ACT + 255) / 256, 256, 0, stream>>>(k_in, kbf, (int)ACT);
    k_f32_to_bf16<<<(ACT + 255) / 256, 256, 0, stream>>>(v_in, vbf, (int)ACT);
    k_f32_to_bf16<<<(WEL + 255) / 256, 256, 0, stream>>>(Wq, wqb, (int)WEL);
    k_f32_to_bf16<<<(WEL + 255) / 256, 256, 0, stream>>>(Wk, wkb, (int)WEL);
    k_f32_to_bf16<<<(WEL + 255) / 256, 256, 0, stream>>>(Wv, wvb, (int)WEL);
    k_f32_to_bf16<<<(WEL + 255) / 256, 256, 0, stream>>>(Wo, wob, (int)WEL);

    // 2) Q/K/V projections (WMMA)
    const int gemm_blocks = (NTOK / 16) * (DMODEL / 64);  // 4096
    k_proj_gemm<<<gemm_blocks, 32, 0, stream>>>(qbf, wqb, qh, 0);
    k_proj_gemm<<<gemm_blocks, 32, 0, stream>>>(kbf, wkb, kh, 1);
    k_proj_gemm<<<gemm_blocks, 32, 0, stream>>>(vbf, wvb, vT, 2);

    // 3) Attention scores + online stats (raw S into attn region)
    const int att_blocks = BATCH * NHEAD * (LQ / 16);     // 4096
    k_att_scores<<<att_blocks, 32, 0, stream>>>(qh, kh, pbias, mask, attn, stats);

    // 4) Normalize attn + P@V (WMMA)
    k_att_pv<<<att_blocks, 32, 0, stream>>>(vT, attn, stats, ho);

    // 5) Output projection + residual (WMMA)
    k_out_gemm<<<gemm_blocks, 32, 0, stream>>>(ho, wob, q_in, pre);

    // 6) LayerNorm
    k_layernorm<<<NTOK, 256, 0, stream>>>(pre, gamma, beta, out);
}
